// SelfAttention_73753178407202
// MI455X (gfx1250) — matrix operand order repaired and checked by reading, not compiled, not run on hardware
//
#include <hip/hip_runtime.h>
#include <hip/hip_bf16.h>

// ---------------------------------------------------------------------------
// Self-attention block for MI455X (gfx1250), wave32 + WMMA bf16 + async LDS,
// software-pipelined (double-buffered) with loop-carried addressing.
// D=1024, H=16, hd=64, seq=2048, batch=2  -> tokens T=4096.
// ---------------------------------------------------------------------------

typedef __attribute__((ext_vector_type(16))) __bf16          v16bf;
typedef __attribute__((ext_vector_type(8)))  float           v8f;
typedef __attribute__((ext_vector_type(8)))  unsigned short  us8;

#define D_MODEL 1024
#define N_SEQ   2048
#define N_HEAD  16
#define HEAD_D  64
#define TOKENS  4096

__device__ __forceinline__ unsigned short f2bf(float f) {
  union { float f; unsigned u; } v; v.f = f;
  return (unsigned short)((v.u + 0x7FFFu + ((v.u >> 16) & 1u)) >> 16);
}

// A-fragment (16x32 bf16): elements 0..15 = K[k0..k0+15]
__device__ __forceinline__ v16bf load_a_frag(const unsigned short* base) {
  union { v16bf v; us8 h[2]; } u;
  u.h[0] = *(const us8*)(base);
  u.h[1] = *(const us8*)(base + 8);
  return u.v;
}

__device__ __forceinline__ v8f wmma_bf16(v16bf a, v16bf b, v8f c) {
  return __builtin_amdgcn_wmma_f32_16x16x32_bf16(false, a, false, b,
                                                 (short)0, c, false, false);
}

// Raw wave-relative LDS byte offset (low 32 bits of the generic pointer).
__device__ __forceinline__ unsigned lds_off(const void* p) {
  return (unsigned)(unsigned long long)p;
}

// Async global->LDS copies.  INST_OFFSET is added to BOTH the global address
// and the LDS address, so contiguous
// multi-b128 copies share one LDS-offset VGPR and one global pointer.
__device__ __forceinline__ void async_copy_64B(unsigned lds, const void* g) {
  asm volatile(
      "global_load_async_to_lds_b128 %0, %1, off\n\t"
      "global_load_async_to_lds_b128 %0, %1, off offset:16\n\t"
      "global_load_async_to_lds_b128 %0, %1, off offset:32\n\t"
      "global_load_async_to_lds_b128 %0, %1, off offset:48"
      :: "v"(lds), "v"(g) : "memory");
}
__device__ __forceinline__ void async_copy_32B(unsigned lds, const void* g) {
  asm volatile(
      "global_load_async_to_lds_b128 %0, %1, off\n\t"
      "global_load_async_to_lds_b128 %0, %1, off offset:16"
      :: "v"(lds), "v"(g) : "memory");
}
__device__ __forceinline__ void async_wait0() {
  asm volatile("s_wait_asynccnt 0x0" ::: "memory");
}

// ---------------------------------------------------------------------------
// Kernel 1: LayerNorm (f32 in) -> bf16 out.  One block (256 thr) per token.
// ---------------------------------------------------------------------------
__global__ void __launch_bounds__(256) ln_kernel(
    const float* __restrict__ x, const float* __restrict__ gamma,
    const float* __restrict__ beta, unsigned short* __restrict__ xn)
{
  const int row = blockIdx.x;
  const int t = threadIdx.x;
  const float* xr = x + (size_t)row * D_MODEL;
  float v[4];
  float s = 0.f, ss = 0.f;
#pragma unroll
  for (int i = 0; i < 4; ++i) {
    v[i] = xr[t + 256 * i];
    s += v[i]; ss += v[i] * v[i];
  }
#pragma unroll
  for (int off = 16; off; off >>= 1) {
    s  += __shfl_xor(s,  off, 32);
    ss += __shfl_xor(ss, off, 32);
  }
  __shared__ float rs[8], rss[8];
  const int w = t >> 5, l = t & 31;
  if (l == 0) { rs[w] = s; rss[w] = ss; }
  __syncthreads();
  if (w == 0) {
    float a  = (l < 8) ? rs[l]  : 0.f;
    float b2 = (l < 8) ? rss[l] : 0.f;
#pragma unroll
    for (int off = 4; off; off >>= 1) {
      a  += __shfl_xor(a,  off, 32);
      b2 += __shfl_xor(b2, off, 32);
    }
    if (l == 0) { rs[0] = a; rss[0] = b2; }
  }
  __syncthreads();
  const float mean = rs[0] * (1.f / D_MODEL);
  const float var  = rss[0] * (1.f / D_MODEL) - mean * mean;
  const float rstd = rsqrtf(var + 1e-6f);
  unsigned short* xo = xn + (size_t)row * D_MODEL;
#pragma unroll
  for (int i = 0; i < 4; ++i) {
    const int c = t + 256 * i;
    xo[c] = f2bf((v[i] - mean) * rstd * gamma[c] + beta[c]);
  }
}

// ---------------------------------------------------------------------------
// Kernel 2: cast + transpose weights: Wt[n][k] = bf16(W[k][n]).
// ---------------------------------------------------------------------------
__global__ void __launch_bounds__(256) cast_transpose_kernel(
    const float* __restrict__ src, unsigned short* __restrict__ dst)
{
  __shared__ unsigned short tile[32][33];
  const int n0 = blockIdx.x * 32, k0 = blockIdx.y * 32;
  const int tx = threadIdx.x & 31, ty = threadIdx.x >> 5;   // 32 x 8
#pragma unroll
  for (int i = 0; i < 32; i += 8)
    tile[ty + i][tx] = f2bf(src[(size_t)(k0 + ty + i) * D_MODEL + n0 + tx]);
  __syncthreads();
#pragma unroll
  for (int i = 0; i < 32; i += 8)
    dst[(size_t)(n0 + ty + i) * D_MODEL + k0 + tx] = tile[tx][ty + i];
}

// ---------------------------------------------------------------------------
// Shared GEMM core: C[128x64] tile of  A[4096x1024] @ Wt^T, software-pipelined
// async LDS staging.  A row-major [m][k], Wt row-major [n][k].
// Each of 8 waves owns 16 M-rows x 64 N-cols -> 4 v8f accumulators.
// ---------------------------------------------------------------------------
struct GemmAcc { v8f a[4]; };

__device__ __forceinline__ void gemm_compute_step(
    const unsigned short* Ac, const unsigned short* Bc,
    int mrow, int l, int k0a, int k0b, GemmAcc& r)
{
  v16bf a0 = load_a_frag(Ac + mrow * 64 + k0a);
  v16bf a1 = load_a_frag(Ac + mrow * 64 + 32 + k0a);
  v16bf bf[8];
#pragma unroll
  for (int ks = 0; ks < 2; ++ks)
#pragma unroll
    for (int nt = 0; nt < 4; ++nt)
      bf[ks * 4 + nt] =
          *(const v16bf*)(Bc + (nt * 16 + (l & 15)) * 64 + ks * 32 + k0b);
#pragma unroll
  for (int nt = 0; nt < 4; ++nt) r.a[nt] = wmma_bf16(a0, bf[nt],     r.a[nt]);
#pragma unroll
  for (int nt = 0; nt < 4; ++nt) r.a[nt] = wmma_bf16(a1, bf[4 + nt], r.a[nt]);
}

__device__ __forceinline__ GemmAcc gemm_128x64(
    const unsigned short* __restrict__ Ag,   // [4096][1024] bf16
    const unsigned short* __restrict__ Wt,   // [1024][1024] bf16, [n][k]
    int m0, int n0, int t,
    unsigned short* As /*2x128*64*/, unsigned short* Bs /*2x64*64*/)
{
  const int w = t >> 5, l = t & 31;
  const int ar = t >> 1, ah = (t & 1) * 32;   // A: 64 B / thread
  const int bn = t >> 2, bg = (t & 3) * 16;   // B: 32 B / thread
  const int mrow = w * 16 + (l & 15);
  const int k0a  = (l < 16) ? 0 : 16;
  const int k0b  = (l < 16) ? 0 : 16;

  GemmAcc r;
#pragma unroll
  for (int nt = 0; nt < 4; ++nt)
    r.a[nt] = (v8f){0.f,0.f,0.f,0.f,0.f,0.f,0.f,0.f};

  // Loop-carried global pointers (advance 64 elements per K-block).
  const unsigned short* aptr = Ag + (size_t)(m0 + ar) * D_MODEL + ah;
  const unsigned short* bptr = Wt + (size_t)(n0 + bn) * D_MODEL + bg;
  // Per-buffer LDS staging offsets (bytes), computed once.
  const unsigned sa0 = lds_off(As + ar * 64 + ah);
  const unsigned sa1 = sa0 + 128 * 64 * 2;
  const unsigned sb0 = lds_off(Bs + bn * 64 + bg);
  const unsigned sb1 = sb0 + 64 * 64 * 2;

  const int nk = D_MODEL / 64;                // 16 K-blocks
  async_copy_64B(sa0, aptr);
  async_copy_32B(sb0, bptr);
  aptr += 64; bptr += 64;
  async_wait0();
  __syncthreads();

#pragma unroll 1
  for (int i = 0; i < nk - 1; ++i) {
    const int cur = i & 1;
    async_copy_64B(cur ? sa0 : sa1, aptr);    // DMA next tile, overlaps WMMAs
    async_copy_32B(cur ? sb0 : sb1, bptr);
    aptr += 64; bptr += 64;
    gemm_compute_step(cur ? As + 128 * 64 : As, cur ? Bs + 64 * 64 : Bs,
                      mrow, l, k0a, k0b, r);
    async_wait0();
    __syncthreads();
  }
  gemm_compute_step(((nk - 1) & 1) ? As + 128 * 64 : As,
                    ((nk - 1) & 1) ? Bs + 64 * 64  : Bs,
                    mrow, l, k0a, k0b, r);
  return r;
}

// ---------------------------------------------------------------------------
// Kernel 3: fused QKV GEMM.  Q,K -> [B,H,N,hd]; V -> transposed [B,H,hd,N].
// ---------------------------------------------------------------------------
__global__ void __launch_bounds__(256) qkv_gemm_kernel(
    const unsigned short* __restrict__ xn,
    const unsigned short* __restrict__ Wqt, const unsigned short* __restrict__ Wkt,
    const unsigned short* __restrict__ Wvt,
    const float* __restrict__ bq, const float* __restrict__ bk,
    const float* __restrict__ bv,
    unsigned short* __restrict__ Q, unsigned short* __restrict__ K,
    unsigned short* __restrict__ V)
{
  const int z = blockIdx.z;
  const unsigned short* Wt = (z == 0) ? Wqt : (z == 1) ? Wkt : Wvt;
  const float* bias        = (z == 0) ? bq  : (z == 1) ? bk  : bv;
  unsigned short* out      = (z == 0) ? Q   : (z == 1) ? K   : V;
  const float scale        = (z == 0) ? 0.125f : 1.0f;   // 1/sqrt(64) into Q

  const int m0 = blockIdx.y * 128;
  const int n0 = blockIdx.x * 64;
  __shared__ __align__(32) unsigned short As[2 * 128 * 64];   // 32 KB
  __shared__ __align__(32) unsigned short Bs[2 * 64 * 64];    // 16 KB

  const int t = threadIdx.x, w = t >> 5, l = t & 31;
  GemmAcc r = gemm_128x64(xn, Wt, m0, n0, t, As, Bs);

#pragma unroll
  for (int nt = 0; nt < 4; ++nt) {
#pragma unroll
    for (int j = 0; j < 8; ++j) {
      const int row = m0 + w * 16 + j + ((l >> 4) << 3);
      const int col = n0 + nt * 16 + (l & 15);
      const float vv = (r.a[nt][j] + bias[col]) * scale;
      const int b  = row >> 11, ns = row & (N_SEQ - 1);
      const int h  = col >> 6,  hd = col & (HEAD_D - 1);
      if (z == 2) {  // V transposed: [b,h,hd,seq]
        out[(((size_t)(b * N_HEAD + h) * HEAD_D) + hd) * N_SEQ + ns] = f2bf(vv);
      } else {       // Q, K: [b,h,seq,hd]
        out[(((size_t)(b * N_HEAD + h) * N_SEQ) + ns) * HEAD_D + hd] = f2bf(vv);
      }
    }
  }
}

// ---------------------------------------------------------------------------
// Kernel 4: flash attention, software-pipelined K/V tiles.
// Block = (head-batch bh, 128 query rows); wave w owns rows [16w,16w+16).
// ---------------------------------------------------------------------------
__global__ void __launch_bounds__(256) attn_kernel(
    const unsigned short* __restrict__ Q, const unsigned short* __restrict__ Kp,
    const unsigned short* __restrict__ Vtg, unsigned short* __restrict__ AO)
{
  const int bh = blockIdx.y;            // b*16 + h
  const int q0 = blockIdx.x * 128;
  const int t = threadIdx.x, w = t >> 5, l = t & 31;

  __shared__ __align__(32) unsigned short Ks[2 * 64 * 64];    // 16 KB [key][hd]
  __shared__ __align__(32) unsigned short Vs[2 * 64 * 64];    // 16 KB [hd][key]
  __shared__ __align__(32) unsigned short Ps[8][16 * 64];     // 16 KB per-wave P

  const size_t base = (size_t)bh * N_SEQ * HEAD_D;

  const int qrow = q0 + w * 16 + (l & 15);
  const int k0a  = (l < 16) ? 0 : 16;
  const int k0b  = (l < 16) ? 0 : 16;
  v16bf aq[2];
  aq[0] = load_a_frag(Q + base + (size_t)qrow * HEAD_D + k0a);
  aq[1] = load_a_frag(Q + base + (size_t)qrow * HEAD_D + 32 + k0a);

  v8f o[4] = {};
  float mrow[8], lrow[8];
#pragma unroll
  for (int j = 0; j < 8; ++j) { mrow[j] = -1e30f; lrow[j] = 0.f; }

  const int lr = t >> 2, lg = (t & 3) * 16;   // tile-load coords (64 x 64)

  // Loop-carried global pointers.
  const unsigned short* kptr = Kp  + base + (size_t)lr * HEAD_D + lg; // +=64*64
  const unsigned short* vptr = Vtg + base + (size_t)lr * N_SEQ  + lg; // +=64
  // Per-buffer LDS staging offsets.
  const unsigned sk0 = lds_off(Ks + lr * 64 + lg);
  const unsigned sk1 = sk0 + 64 * 64 * 2;
  const unsigned sv0 = lds_off(Vs + lr * 64 + lg);
  const unsigned sv1 = sv0 + 64 * 64 * 2;

  const int ntile = N_SEQ / 64;               // 32 key tiles
  async_copy_32B(sk0, kptr);
  async_copy_32B(sv0, vptr);
  kptr += 64 * HEAD_D; vptr += 64;
  async_wait0();
  __syncthreads();

#pragma unroll 1
  for (int i = 0; i < ntile; ++i) {
    const int cur = i & 1;
    if (i + 1 < ntile) {
      async_copy_32B(cur ? sk0 : sk1, kptr);  // DMA next tile, overlaps compute
      async_copy_32B(cur ? sv0 : sv1, vptr);
      kptr += 64 * HEAD_D; vptr += 64;
    }
    const unsigned short* Kc = cur ? Ks + 64 * 64 : Ks;
    const unsigned short* Vc = cur ? Vs + 64 * 64 : Vs;

    // S = Q K^T : preload 8 B-fragments, then 8 WMMAs.
    v16bf bk8[8];
#pragma unroll
    for (int ks = 0; ks < 2; ++ks)
#pragma unroll
      for (int nt = 0; nt < 4; ++nt)
        bk8[ks * 4 + nt] =
            *(const v16bf*)(Kc + (nt * 16 + (l & 15)) * 64 + ks * 32 + k0b);
    v8f s[4] = {};
#pragma unroll
    for (int nt = 0; nt < 4; ++nt) s[nt] = wmma_bf16(aq[0], bk8[nt],     s[nt]);
#pragma unroll
    for (int nt = 0; nt < 4; ++nt) s[nt] = wmma_bf16(aq[1], bk8[4 + nt], s[nt]);

    // Online softmax; rows live in 16-lane groups (row = j + 8*(l>=16)).
    float p[4][8];
#pragma unroll
    for (int j = 0; j < 8; ++j) {
      float tm = fmaxf(fmaxf(s[0][j], s[1][j]), fmaxf(s[2][j], s[3][j]));
#pragma unroll
      for (int off = 8; off; off >>= 1) tm = fmaxf(tm, __shfl_xor(tm, off, 16));
      const float mn   = fmaxf(mrow[j], tm);
      const float corr = __expf(mrow[j] - mn);
      mrow[j] = mn;
      float rsum = 0.f;
#pragma unroll
      for (int nt = 0; nt < 4; ++nt) {
        const float e = __expf(s[nt][j] - mn);
        p[nt][j] = e; rsum += e;
      }
#pragma unroll
      for (int off = 8; off; off >>= 1) rsum += __shfl_xor(rsum, off, 16);
      lrow[j] = lrow[j] * corr + rsum;
#pragma unroll
      for (int nt = 0; nt < 4; ++nt) o[nt][j] *= corr;
    }

    // C-layout P -> per-wave LDS (row-major 16x64) -> A fragments.
#pragma unroll
    for (int nt = 0; nt < 4; ++nt)
#pragma unroll
      for (int j = 0; j < 8; ++j)
        Ps[w][(j + ((l >> 4) << 3)) * 64 + nt * 16 + (l & 15)] = f2bf(p[nt][j]);

    // P V : preload fragments, then 8 WMMAs.
    v16bf ap0 = load_a_frag(&Ps[w][(l & 15) * 64 + k0a]);
    v16bf ap1 = load_a_frag(&Ps[w][(l & 15) * 64 + 32 + k0a]);
    v16bf bv8[8];
#pragma unroll
    for (int ks = 0; ks < 2; ++ks)
#pragma unroll
      for (int nt = 0; nt < 4; ++nt)
        bv8[ks * 4 + nt] =
            *(const v16bf*)(Vc + (nt * 16 + (l & 15)) * 64 + ks * 32 + k0b);
#pragma unroll
    for (int nt = 0; nt < 4; ++nt) o[nt] = wmma_bf16(ap0, bv8[nt],     o[nt]);
#pragma unroll
    for (int nt = 0; nt < 4; ++nt) o[nt] = wmma_bf16(ap1, bv8[4 + nt], o[nt]);

    if (i + 1 < ntile) async_wait0();
    __syncthreads();
  }

  // Normalize and scatter to [token][D] bf16.
#pragma unroll
  for (int nt = 0; nt < 4; ++nt) {
#pragma unroll
    for (int j = 0; j < 8; ++j) {
      const int row   = q0 + w * 16 + j + ((l >> 4) << 3);
      const int col   = (bh & (N_HEAD - 1)) * HEAD_D + nt * 16 + (l & 15);
      const int token = (bh >> 4) * N_SEQ + row;
      const float vv  = o[nt][j] / lrow[j];
      AO[(size_t)token * D_MODEL + col] = f2bf(vv);
    }
  }
}

// ---------------------------------------------------------------------------
// Kernel 5: output projection + bias + residual -> f32 out.
// ---------------------------------------------------------------------------
__global__ void __launch_bounds__(256) oproj_kernel(
    const unsigned short* __restrict__ A,    // attn out bf16 [4096][1024]
    const unsigned short* __restrict__ Wt,   // Wo^T bf16 [n][k]
    const float* __restrict__ bo,
    const float* __restrict__ x,             // residual
    float* __restrict__ out)
{
  const int m0 = blockIdx.y * 128;
  const int n0 = blockIdx.x * 64;
  __shared__ __align__(32) unsigned short As[2 * 128 * 64];
  __shared__ __align__(32) unsigned short Bs[2 * 64 * 64];

  const int t = threadIdx.x, w = t >> 5, l = t & 31;
  GemmAcc r = gemm_128x64(A, Wt, m0, n0, t, As, Bs);

#pragma unroll
  for (int nt = 0; nt < 4; ++nt) {
#pragma unroll
    for (int j = 0; j < 8; ++j) {
      const int row = m0 + w * 16 + j + ((l >> 4) << 3);
      const int col = n0 + nt * 16 + (l & 15);
      out[(size_t)row * D_MODEL + col] =
          r.a[nt][j] + bo[col] + x[(size_t)row * D_MODEL + col];
    }
  }
}

// ---------------------------------------------------------------------------
// Launcher
// ---------------------------------------------------------------------------
extern "C" void kernel_launch(void* const* d_in, const int* in_sizes, int n_in,
                              void* d_out, int out_size, void* d_ws, size_t ws_size,
                              hipStream_t stream)
{
  (void)in_sizes; (void)n_in; (void)out_size; (void)ws_size;

  const float* x  = (const float*)d_in[0];
  const float* Wq = (const float*)d_in[1];
  const float* bq = (const float*)d_in[2];
  const float* Wk = (const float*)d_in[3];
  const float* bk = (const float*)d_in[4];
  const float* Wv = (const float*)d_in[5];
  const float* bv = (const float*)d_in[6];
  const float* Wo = (const float*)d_in[7];
  const float* bo = (const float*)d_in[8];
  const float* g  = (const float*)d_in[9];
  const float* be = (const float*)d_in[10];

  char* ws = (char*)d_ws;
  unsigned short* xn   = (unsigned short*)(ws);                      // 8 MB
  unsigned short* Wqt  = (unsigned short*)(ws + ( 8ull << 20));      // 2 MB
  unsigned short* Wkt  = (unsigned short*)(ws + (10ull << 20));      // 2 MB
  unsigned short* Wvt  = (unsigned short*)(ws + (12ull << 20));      // 2 MB
  unsigned short* Wot  = (unsigned short*)(ws + (14ull << 20));      // 2 MB
  unsigned short* Qb   = (unsigned short*)(ws + (16ull << 20));      // 8 MB
  unsigned short* Kb   = (unsigned short*)(ws + (24ull << 20));      // 8 MB
  unsigned short* Vtb  = (unsigned short*)(ws + (32ull << 20));      // 8 MB
  unsigned short* AOb  = (unsigned short*)(ws + (40ull << 20));      // 8 MB

  ln_kernel<<<TOKENS, 256, 0, stream>>>(x, g, be, xn);
  cast_transpose_kernel<<<dim3(32, 32), 256, 0, stream>>>(Wq, Wqt);
  cast_transpose_kernel<<<dim3(32, 32), 256, 0, stream>>>(Wk, Wkt);
  cast_transpose_kernel<<<dim3(32, 32), 256, 0, stream>>>(Wv, Wvt);
  cast_transpose_kernel<<<dim3(32, 32), 256, 0, stream>>>(Wo, Wot);
  qkv_gemm_kernel<<<dim3(16, 32, 3), 256, 0, stream>>>(
      xn, Wqt, Wkt, Wvt, bq, bk, bv, Qb, Kb, Vtb);
  attn_kernel<<<dim3(16, 32), 256, 0, stream>>>(Qb, Kb, Vtb, AOb);
  oproj_kernel<<<dim3(16, 32), 256, 0, stream>>>(AOb, Wot, bo, x, (float*)d_out);
}
